// BiMambaBlock_62216896250344
// MI455X (gfx1250) — compile-verified, matrix operand order checked by reading, not run on hardware
//
#include <hip/hip_runtime.h>
#include <hip/hip_bf16.h>
#include <math.h>

typedef __attribute__((ext_vector_type(16))) _Float16 v16h;
typedef __attribute__((ext_vector_type(8)))  _Float16 v8h;
typedef __attribute__((ext_vector_type(8)))  float    v8f;

#define D_MODEL 512
#define D_INNER 1024
#define D_STATE 16
#define DT_RANK 32
#define SEQ_L   2048
#define BATCH   8
#define NTOK    (BATCH*SEQ_L)   /* 16384 tokens */

// ---------------------------------------------------------------------------
// GEMM: C[M x N] (f32) = A[M x K] (f32, lda) * W[K x N] (f32 row-major, ldw)
// f16 WMMA compute, f32 accumulate, f32->f16 conversion during LDS staging.
//
// LDS layout is fragment-native: each 32-wide k row is stored with the middle
// two k-octets swapped (position order k = 0-7,16-23,8-15,24-31), matching the
// ISA 7.12.2 16-bit A/B fragment layout. B is stored transposed (WsT[n][k]).
// A lane loads a whole fragment as two aligned 16-byte LDS reads
// (ds_load_b128 x2). Row stride 40 halves (80 B) keeps 16-B alignment and a
// conflict-free bank spread. Double-buffered: next tile is staged between the
// fragment loads and the WMMAs. W-column addresses are materialized into
// distinct registers before loading so the 8 b32 loads clause (no xcnt WAR
// waits on a reused address register).
//
// Requires M%64==0, N%64==0, K%32==0 (true for all four GEMMs here).
// Block = 256 threads = 8 waves; block tile 64x64; wave tile 16x32 (2 WMMAs).
// ---------------------------------------------------------------------------
__global__ __launch_bounds__(256)
void gemm_wmma_f16(const float* __restrict__ A, int lda,
                   const float* __restrict__ W, int ldw,
                   float* __restrict__ C, int ldc,
                   int K, int revL)
{
    __shared__ __attribute__((aligned(16))) _Float16 AsF[2][64][40];
    __shared__ __attribute__((aligned(16))) _Float16 WsT[2][64][40];

    const int tid  = threadIdx.x;
    const int wave = tid >> 5;
    const int lane = tid & 31;
    const int lh   = lane >> 4;        // lane half (0/1)
    const int ln   = lane & 15;
    const int bm   = blockIdx.y * 64;
    const int bn   = blockIdx.x * 64;
    const int wm   = (wave & 3) * 16;  // wave row offset in block tile
    const int wn   = (wave >> 2) * 32; // wave col offset in block tile

    // Staging role: one (row, k-octet) task per thread, one pass per tile.
    const int sr = tid & 63;                        // A row / W column (n)
    const int so = tid >> 6;                        // k-octet 0..3
    const int sp = ((so & 1) << 1) | (so >> 1);     // permuted octet {0,2,1,3}

    int gra = bm + sr;                              // branchless L-flip of A rows
    if (revL) { int l = gra & (SEQ_L - 1); gra += (SEQ_L - 1) - 2 * l; }
    const float* __restrict__ Aseg = A + (size_t)gra * lda + so * 8;
    const float* __restrict__ Wseg = W + (size_t)(so * 8) * ldw + (bn + sr);
    const size_t wst = (size_t)ldw;

    v8f acc0 = {}; v8f acc1 = {};

    auto stage = [&](int k0, int buf) {
        // A: 8 contiguous f32 per lane -> two global_load_b128; one ds_store_b128.
        v8h ah;
        const float* __restrict__ Ak = Aseg + k0;
#pragma unroll
        for (int j = 0; j < 8; ++j) ah[j] = (_Float16)Ak[j];
        *(v8h*)&AsF[buf][sr][sp * 8] = ah;
        // W: walk k. Materialize 8 independent addresses first, then load, so
        // the loads clause instead of chaining on one address register.
        const float* __restrict__ Wk = Wseg + (size_t)k0 * wst;
        const float* p[8];
#pragma unroll
        for (int j = 0; j < 8; ++j) p[j] = Wk + (size_t)j * wst;
        float wv[8];
#pragma unroll
        for (int j = 0; j < 8; ++j) wv[j] = *p[j];
        v8h wh;
#pragma unroll
        for (int j = 0; j < 8; ++j) wh[j] = (_Float16)wv[j];
        *(v8h*)&WsT[buf][sr][sp * 8] = wh;
    };

    stage(0, 0);
    int buf = 0;
    for (int k0 = 0; k0 < K; k0 += 32) {
        __syncthreads();
        // Fragment loads: two aligned 16-B LDS reads each (ds_load_b128 x2)
        v8h a_lo = *(const v8h*)&AsF[buf][wm + ln][lh * 16];
        v8h a_hi = *(const v8h*)&AsF[buf][wm + ln][lh * 16 + 8];
        v8h b0lo = *(const v8h*)&WsT[buf][wn + ln][lh * 16];
        v8h b0hi = *(const v8h*)&WsT[buf][wn + ln][lh * 16 + 8];
        v8h b1lo = *(const v8h*)&WsT[buf][wn + 16 + ln][lh * 16];
        v8h b1hi = *(const v8h*)&WsT[buf][wn + 16 + ln][lh * 16 + 8];
        // Stage next tile into the other buffer (overlaps with WMMA below)
        if (k0 + 32 < K) stage(k0 + 32, buf ^ 1);
        v16h af  = __builtin_shufflevector(a_lo, a_hi,
                     0,1,2,3,4,5,6,7,8,9,10,11,12,13,14,15);
        v16h bf0 = __builtin_shufflevector(b0lo, b0hi,
                     0,1,2,3,4,5,6,7,8,9,10,11,12,13,14,15);
        v16h bf1 = __builtin_shufflevector(b1lo, b1hi,
                     0,1,2,3,4,5,6,7,8,9,10,11,12,13,14,15);
        acc0 = __builtin_amdgcn_wmma_f32_16x16x32_f16(false, af, false, bf0,
                                                      (short)0, acc0, false, false);
        acc1 = __builtin_amdgcn_wmma_f32_16x16x32_f16(false, af, false, bf1,
                                                      (short)0, acc1, false, false);
        buf ^= 1;
    }

    // D layout: VGPR r -> row r (lanes 0-15) / row r+8 (lanes 16-31); lane%16 = col
#pragma unroll
    for (int r = 0; r < 8; ++r) {
        int gr = bm + wm + r + 8 * lh;
        int gc = bn + wn + ln;
        C[(size_t)gr * ldc + gc]      = acc0[r];
        C[(size_t)gr * ldc + gc + 16] = acc1[r];
    }
}

// ---------------------------------------------------------------------------
// Fused causal depthwise conv (k=4) + bias + SiLU.
// Reads u = xz[:, 0:1024] (row stride 2048), writes dense u [NTOK x 1024].
// ---------------------------------------------------------------------------
__global__ __launch_bounds__(256)
void conv_silu_kernel(const float* __restrict__ xz, const float* __restrict__ cw,
                      const float* __restrict__ cb, float* __restrict__ u)
{
    int idx = blockIdx.x * 256 + threadIdx.x;   // over NTOK*1024
    int c = idx & (D_INNER - 1);
    int m = idx >> 10;
    int l = m & (SEQ_L - 1);
    float acc = cb[c];
    const float* w = cw + c * 4;
#pragma unroll
    for (int k = 0; k < 4; ++k) {
        int ls = l - 3 + k;
        if (ls >= 0)
            acc += w[k] * xz[(size_t)(m - l + ls) * 2048 + c];
    }
    u[idx] = acc / (1.f + __expf(-acc));        // SiLU
}

// dt = softplus(dtmat + bias)
__global__ __launch_bounds__(256)
void dt_softplus_kernel(float* __restrict__ dt, const float* __restrict__ bdt)
{
    int idx = blockIdx.x * 256 + threadIdx.x;
    int c = idx & (D_INNER - 1);
    float x = dt[idx] + bdt[c];
    dt[idx] = (x > 20.f) ? x : log1pf(__expf(x));
}

// ---------------------------------------------------------------------------
// Selective scan, fused with +u*D and *silu(z). In-place over u (uy).
// Grid: (BATCH, D_INNER/256). Each thread owns one channel: h[16] + A row in
// registers; B_t/C_t (shared by all channels of a token) double-buffered in
// LDS -> one barrier per timestep. Streamed operands use non-temporal hints.
// ---------------------------------------------------------------------------
__global__ __launch_bounds__(256)
void scan_kernel(const float* __restrict__ xdbl, const float* __restrict__ dtv,
                 const float* __restrict__ xz, const float* __restrict__ Alog,
                 const float* __restrict__ Dp, float* __restrict__ uy)
{
    int b  = blockIdx.x;
    int ci = blockIdx.y * 256 + threadIdx.x;

    float Ar[D_STATE], h[D_STATE];
#pragma unroll
    for (int n = 0; n < D_STATE; ++n) {
        Ar[n] = -__expf(Alog[ci * D_STATE + n]);
        h[n] = 0.f;
    }
    float Dc = Dp[ci];

    __shared__ float BC[2][32];    // [0:16)=B_t, [16:32)=C_t, double-buffered
    size_t row0 = (size_t)b * SEQ_L;
    if (threadIdx.x < 32)
        BC[0][threadIdx.x] = xdbl[row0 * 64 + DT_RANK + threadIdx.x];
    __syncthreads();

    for (int l = 0; l < SEQ_L; ++l) {
        size_t row = row0 + l;
        int cur = l & 1;
        // Prefetch next step's B/C into the other buffer (no barrier needed
        // until end of iteration).
        if (l + 1 < SEQ_L && threadIdx.x < 32)
            BC[cur ^ 1][threadIdx.x] = xdbl[(row + 1) * 64 + DT_RANK + threadIdx.x];

        float ut  = __builtin_nontemporal_load(&uy [row * 1024 + ci]);
        float dtt = __builtin_nontemporal_load(&dtv[row * 1024 + ci]);
        float zt  = __builtin_nontemporal_load(&xz [row * 2048 + 1024 + ci]);
        float du  = dtt * ut;
        float y = 0.f;
#pragma unroll
        for (int n = 0; n < D_STATE; ++n) {
            float dA = __expf(dtt * Ar[n]);
            h[n] = h[n] * dA + du * BC[cur][n];
            y = fmaf(h[n], BC[cur][16 + n], y);
        }
        y += ut * Dc;
        float sz = zt / (1.f + __expf(-zt));
        __builtin_nontemporal_store(y * sz, &uy[row * 1024 + ci]);
        __syncthreads();   // next buffer ready + this buffer safe to overwrite
    }
}

// s (+)= dirout, with L-flip for the backward direction
__global__ __launch_bounds__(256)
void accum_kernel(const float* __restrict__ dirout, float* __restrict__ s,
                  int rev, int add)
{
    int idx = blockIdx.x * 256 + threadIdx.x;   // over NTOK*512
    int j = idx & (D_MODEL - 1);
    int m = idx >> 9;
    int l = m & (SEQ_L - 1);
    int ls = rev ? (SEQ_L - 1 - l) : l;
    float v = dirout[(size_t)(m - l + ls) * D_MODEL + j];
    if (add) s[idx] += v; else s[idx] = v;
}

// LayerNorm over d_model=512; one block per token, 2 elems/thread
__global__ __launch_bounds__(256)
void ln_kernel(const float* __restrict__ s, const float* __restrict__ g,
               const float* __restrict__ bb, float* __restrict__ out)
{
    __shared__ float sm[256];
    int m = blockIdx.x, tid = threadIdx.x;
    const float* row = s + (size_t)m * D_MODEL;
    float v0 = row[tid], v1 = row[tid + 256];

    sm[tid] = v0 + v1; __syncthreads();
    for (int off = 128; off > 0; off >>= 1) {
        if (tid < off) sm[tid] += sm[tid + off];
        __syncthreads();
    }
    float mu = sm[0] * (1.f / 512.f);
    __syncthreads();

    float d0 = v0 - mu, d1 = v1 - mu;
    sm[tid] = d0 * d0 + d1 * d1; __syncthreads();
    for (int off = 128; off > 0; off >>= 1) {
        if (tid < off) sm[tid] += sm[tid + off];
        __syncthreads();
    }
    float inv = rsqrtf(sm[0] * (1.f / 512.f) + 1e-5f);

    out[(size_t)m * D_MODEL + tid]       = d0 * inv * g[tid]       + bb[tid];
    out[(size_t)m * D_MODEL + tid + 256] = d1 * inv * g[tid + 256] + bb[tid + 256];
}

// ---------------------------------------------------------------------------
extern "C" void kernel_launch(void* const* d_in, const int* in_sizes, int n_in,
                              void* d_out, int out_size, void* d_ws, size_t ws_size,
                              hipStream_t stream)
{
    (void)in_sizes; (void)n_in; (void)out_size; (void)ws_size;
    const float* x    = (const float*)d_in[0];
    const float* ln_g = (const float*)d_in[19];
    const float* ln_b = (const float*)d_in[20];

    // Workspace layout (bytes): XZ | UCV | XDBL | DT | S   (~292 MB total)
    char* ws = (char*)d_ws;
    float* XZ     = (float*)(ws);                                          // NTOK x 2048
    float* UCV    = (float*)(ws + (size_t)NTOK*2048*4);                    // NTOK x 1024 (u -> gated y, in place)
    float* XDBL   = (float*)(ws + (size_t)NTOK*2048*4 + (size_t)NTOK*1024*4);        // NTOK x 64
    float* DT     = (float*)(ws + (size_t)NTOK*2048*4 + (size_t)NTOK*1024*4
                                + (size_t)NTOK*64*4);                      // NTOK x 1024
    float* S      = (float*)(ws + (size_t)NTOK*2048*4 + (size_t)NTOK*1024*4
                                + (size_t)NTOK*64*4 + (size_t)NTOK*1024*4);// NTOK x 512
    float* DIROUT = XZ;  // xz fully consumed by scan; reuse its head for per-dir output

    const dim3 blk(256);
    const int MT = NTOK / 64;   // 256 block rows

    for (int dir = 0; dir < 2; ++dir) {
        int base = 1 + dir * 9;
        const float* Wxz  = (const float*)d_in[base + 0];
        const float* cw   = (const float*)d_in[base + 1];
        const float* cb   = (const float*)d_in[base + 2];
        const float* Wx   = (const float*)d_in[base + 3];
        const float* Wdt  = (const float*)d_in[base + 4];
        const float* bdt  = (const float*)d_in[base + 5];
        const float* Alog = (const float*)d_in[base + 6];
        const float* Dp   = (const float*)d_in[base + 7];
        const float* Wout = (const float*)d_in[base + 8];

        // xz = (flip?)x @ Wxz    [NTOK x 2048]
        gemm_wmma_f16<<<dim3(2048/64, MT), blk, 0, stream>>>(
            x, D_MODEL, Wxz, 2048, XZ, 2048, D_MODEL, dir);
        // u = silu(causal_conv4(xz[:, :1024]) + b)
        conv_silu_kernel<<<NTOK*1024/256, blk, 0, stream>>>(XZ, cw, cb, UCV);
        // xdbl = u @ Wx          [NTOK x 64] = [dt(32) | B(16) | C(16)]
        gemm_wmma_f16<<<dim3(64/64, MT), blk, 0, stream>>>(
            UCV, 1024, Wx, 64, XDBL, 64, 1024, 0);
        // dt = xdbl[:, :32] @ Wdt  [NTOK x 1024]
        gemm_wmma_f16<<<dim3(1024/64, MT), blk, 0, stream>>>(
            XDBL, 64, Wdt, 1024, DT, 1024, DT_RANK, 0);
        dt_softplus_kernel<<<NTOK*1024/256, blk, 0, stream>>>(DT, bdt);
        // selective scan + (+u*D) + *silu(z), in place over UCV
        scan_kernel<<<dim3(BATCH, D_INNER/256), blk, 0, stream>>>(
            XDBL, DT, XZ, Alog, Dp, UCV);
        // dirout = y @ Wout      [NTOK x 512]
        gemm_wmma_f16<<<dim3(512/64, MT), blk, 0, stream>>>(
            UCV, 1024, Wout, 512, DIROUT, 512, 1024, 0);
        // s (+)= (flip?)dirout
        accum_kernel<<<NTOK*512/256, blk, 0, stream>>>(DIROUT, S, dir, dir);
    }

    ln_kernel<<<NTOK, blk, 0, stream>>>(S, ln_g, ln_b, (float*)d_out);
}
